// MultiHeadAttention_60739427501016
// MI455X (gfx1250) — compile-verified, matrix operand order checked by reading, not run on hardware
//
#include <hip/hip_runtime.h>

typedef __attribute__((ext_vector_type(16))) _Float16 v16h;
typedef __attribute__((ext_vector_type(8)))  _Float16 v8h;
typedef __attribute__((ext_vector_type(4)))  _Float16 v4h;
typedef __attribute__((ext_vector_type(8)))  float    v8f;
typedef __attribute__((ext_vector_type(4)))  unsigned int v4u;
typedef __attribute__((ext_vector_type(8)))  int      v8i;
typedef __attribute__((ext_vector_type(4)))  int      v4i;

union AFrag { v16h v; v8h h[2]; };

// ds_swizzle xor-mode: bit15=0, and_mask=0x1f, or_mask=0, xor_mask=m
#define SWZF(x, m) __int_as_float(__builtin_amdgcn_ds_swizzle(__float_as_int(x), (0x1f | ((m) << 10))))

// ---------------------------------------------------------------------------
// TDM: 2D tile (tile_k x tile_rows) of f16, row stride kstride (elements),
// into LDS at lds_off with +4-DWORD pad every 16 DWORDs (=> 40-half rows).
// D# per CDNA5 ISA 8.3/8.4 (group0 128b, group1 256b; groups 2/3 zero => 2D).
// This toolchain exposes the 6-arg builtin (extra int32x8 group before cpol).
// ---------------------------------------------------------------------------
__device__ inline void tdm_load_tile_f16(unsigned int lds_off, const _Float16* gp,
                                         int tile_rows, int tensor_rows, int kstride) {
  unsigned long long ga = (unsigned long long)(uintptr_t)gp;
  v4u g0;
  g0[0] = 1u;                                        // count=1, no gather
  g0[1] = lds_off;                                   // lds_addr
  g0[2] = (unsigned int)ga;                          // global_addr[31:0]
  g0[3] = ((unsigned int)(ga >> 32) & 0x01FFFFFFu)   // global_addr[56:32]
        | (2u << 30);                                // type=2 (image)
  v8i g1;
  g1[0] = (int)((1u << 16)                           // data_size = 2 bytes
        | (1u << 20)                                 // pad_enable
        | (3u << 22)                                 // pad_interval: 16 DWORDs
        | (3u << 25));                               // pad_amount:   4 DWORDs
  g1[1] = (int)(((unsigned)kstride & 0xFFFFu) << 16);            // tensor_dim0 lo16
  g1[2] = (int)((((unsigned)kstride >> 16) & 0xFFFFu)
        | (((unsigned)tensor_rows & 0xFFFFu) << 16));            // dim0 hi16 | dim1 lo16
  g1[3] = (int)((((unsigned)tensor_rows >> 16) & 0xFFFFu)
        | (32u << 16));                                          // dim1 hi16 | tile_dim0=32
  g1[4] = (int)((unsigned)tile_rows & 0xFFFFu);                  // tile_dim1 | tile_dim2=0
  g1[5] = kstride;                                               // tensor_dim0_stride lo32
  g1[6] = 0;                                                     // stride hi16 | dim1_stride lo16
  g1[7] = 0;
  v4i z0 = {}, z1 = {};
  v8i z2 = {};
  __builtin_amdgcn_tensor_load_to_lds(g0, g1, z0, z1, z2, 0);
}

// ---------------------------------------------------------------------------
// x (f32) -> xh (f16), vectorized
// ---------------------------------------------------------------------------
__global__ __launch_bounds__(256)
void cvt_f32_f16_kernel(const float* __restrict__ x, _Float16* __restrict__ xh, int n) {
  int i = (blockIdx.x * 256 + threadIdx.x) * 4;
  if (i < n) {
    float4 f = *reinterpret_cast<const float4*>(x + i);
    v4h h;
    h[0] = (_Float16)f.x; h[1] = (_Float16)f.y;
    h[2] = (_Float16)f.z; h[3] = (_Float16)f.w;
    *reinterpret_cast<v4h*>(xh + i) = h;
  }
}

// ---------------------------------------------------------------------------
// Wt[n][k] = (f16)W[k][n]   (32x32 tiles via LDS)
// ---------------------------------------------------------------------------
__global__ __launch_bounds__(256)
void transpose_cvt_kernel(const float* __restrict__ W, _Float16* __restrict__ Wt,
                          int K, int N) {
  __shared__ float tile[32][33];
  int n0 = blockIdx.x * 32, k0 = blockIdx.y * 32;
  int tx = threadIdx.x, ty = threadIdx.y;   // (32, 8)
  #pragma unroll
  for (int i = 0; i < 32; i += 8)
    tile[ty + i][tx] = W[(size_t)(k0 + ty + i) * N + n0 + tx];
  __syncthreads();
  #pragma unroll
  for (int i = 0; i < 32; i += 8)
    Wt[(size_t)(n0 + ty + i) * K + k0 + tx] = (_Float16)tile[tx][ty + i];
}

// ---------------------------------------------------------------------------
// C[M,N] = A[m][k](f16) @ Bt[n][k](f16)^T + bias.  TDM-staged LDS tiles.
// Block tile 128x256, 8 waves (2x4), wave tile 64x64 (16 WMMAs / K-step).
// ---------------------------------------------------------------------------
template <typename TOut>
__global__ __launch_bounds__(256)
void gemm_tdm_kernel(const _Float16* __restrict__ A, const _Float16* __restrict__ Bt,
                     const float* __restrict__ bias, TOut* __restrict__ C,
                     int M, int N, int K) {
  __shared__ _Float16 As[128][40];   // TDM pad => 40-half row stride
  __shared__ _Float16 Bs[256][40];
  const int t    = threadIdx.x;
  const int wave = t >> 5, lane = t & 31;
  const int wm   = wave >> 2, wn = wave & 3;       // 2 x 4 wave grid
  const int m0   = blockIdx.y * 128, n0 = blockIdx.x * 256;
  const int l15  = lane & 15, kh = lane >> 4;

  const unsigned int ldsA = (unsigned int)(uintptr_t)&As[0][0];
  const unsigned int ldsB = (unsigned int)(uintptr_t)&Bs[0][0];

  v8f acc[4][4] = {};

  for (int k0 = 0; k0 < K; k0 += 32) {
    __syncthreads();                 // all waves done reading previous tiles
    if (t < 32) {                    // wave 0 drives the Tensor Data Mover
      tdm_load_tile_f16(ldsA, A + (size_t)m0 * K + k0, 128, M, K);
      tdm_load_tile_f16(ldsB, Bt + (size_t)n0 * K + k0, 256, N, K);
      __builtin_amdgcn_s_wait_tensorcnt(0);
    }
    __syncthreads();                 // publish TDM-written tiles

    AFrag a[4], b[4];
    #pragma unroll
    for (int i = 0; i < 4; ++i) {
      int row = wm * 64 + i * 16 + l15;
      a[i].h[0] = *reinterpret_cast<const v8h*>(&As[row][kh * 8]);
      a[i].h[1] = *reinterpret_cast<const v8h*>(&As[row][16 + kh * 8]);
    }
    #pragma unroll
    for (int j = 0; j < 4; ++j) {
      int col = wn * 64 + j * 16 + l15;
      b[j].h[0] = *reinterpret_cast<const v8h*>(&Bs[col][kh * 8]);
      b[j].h[1] = *reinterpret_cast<const v8h*>(&Bs[col][16 + kh * 8]);
    }
    #pragma unroll
    for (int i = 0; i < 4; ++i)
      #pragma unroll
      for (int j = 0; j < 4; ++j)
        acc[i][j] = __builtin_amdgcn_wmma_f32_16x16x32_f16(
            false, a[i].v, false, b[j].v, (short)0, acc[i][j], false, false);
  }

  #pragma unroll
  for (int j = 0; j < 4; ++j) {
    int col = n0 + wn * 64 + j * 16 + l15;
    float bval = bias[col];
    #pragma unroll
    for (int i = 0; i < 4; ++i) {
      #pragma unroll
      for (int r = 0; r < 8; ++r) {
        int row = m0 + wm * 64 + i * 16 + r + 8 * kh;
        C[(size_t)row * N + col] = (TOut)(acc[i][j][r] + bval);
      }
    }
  }
}

// ---------------------------------------------------------------------------
// Flash attention: block = (qt, h, b), 128 threads, 4 waves x 16 query rows.
// ---------------------------------------------------------------------------
__global__ __launch_bounds__(128)
void flash_attn_kernel(const _Float16* __restrict__ Q, const _Float16* __restrict__ K,
                       const _Float16* __restrict__ V, _Float16* __restrict__ Y) {
  __shared__ _Float16 Vt[64][72];  // [hs][key]
  __shared__ _Float16 Ps[64][72];  // [qrow][key], wave-private 16-row bands
  const int t    = threadIdx.x;
  const int wave = t >> 5, lane = t & 31;
  const int l15  = lane & 15, kh = lane >> 4;
  const int q0   = blockIdx.x * 64;
  const int h    = blockIdx.y, b = blockIdx.z;
  const int T = 2048, EMB = 1024, HS = 64;
  const size_t bOff = (size_t)b * T * EMB + (size_t)h * HS;
  const float scale = 0.125f;  // 1/sqrt(64)

  AFrag aq[2];
  {
    int row = q0 + wave * 16 + l15;
    const _Float16* qp = Q + bOff + (size_t)row * EMB;
    #pragma unroll
    for (int s = 0; s < 2; ++s) {
      aq[s].h[0] = *reinterpret_cast<const v8h*>(qp + s * 32 + kh * 8);
      aq[s].h[1] = *reinterpret_cast<const v8h*>(qp + s * 32 + 16 + kh * 8);
    }
  }

  float mi[8], li[8];
  v8f oacc[4] = {};
  #pragma unroll
  for (int r = 0; r < 8; ++r) { mi[r] = -3.0e38f; li[r] = 0.0f; }

  for (int kt0 = 0; kt0 <= q0; kt0 += 64) {
    __syncthreads();
    // V tile (64 keys x 64 hs) -> Vt[hs][key]
    {
      int key = t >> 1, hs0 = (t & 1) * 32;
      const _Float16* vp = V + bOff + (size_t)(kt0 + key) * EMB;
      #pragma unroll
      for (int c = 0; c < 4; ++c) {
        v8h vv = *reinterpret_cast<const v8h*>(vp + hs0 + c * 8);
        #pragma unroll
        for (int u = 0; u < 8; ++u) Vt[hs0 + c * 8 + u][key] = vv[u];
      }
    }
    __syncthreads();

    // S = Q @ K^T : K fragments straight from global ([pos][hs] contiguous/lane)
    v8f sacc[4] = {};
    #pragma unroll
    for (int j = 0; j < 4; ++j) {
      int key = kt0 + j * 16 + l15;
      const _Float16* kp = K + bOff + (size_t)key * EMB;
      AFrag bk0, bk1;
      bk0.h[0] = *reinterpret_cast<const v8h*>(kp + kh * 8);
      bk0.h[1] = *reinterpret_cast<const v8h*>(kp + 16 + kh * 8);
      bk1.h[0] = *reinterpret_cast<const v8h*>(kp + 32 + kh * 8);
      bk1.h[1] = *reinterpret_cast<const v8h*>(kp + 48 + kh * 8);
      sacc[j] = __builtin_amdgcn_wmma_f32_16x16x32_f16(
          false, aq[0].v, false, bk0.v, (short)0, sacc[j], false, false);
      sacc[j] = __builtin_amdgcn_wmma_f32_16x16x32_f16(
          false, aq[1].v, false, bk1.v, (short)0, sacc[j], false, false);
    }

    if (kt0 == q0) {  // causal mask, diagonal tile only
      #pragma unroll
      for (int j = 0; j < 4; ++j) {
        int key = kt0 + j * 16 + l15;
        #pragma unroll
        for (int r = 0; r < 8; ++r) {
          int qrow = q0 + wave * 16 + r + 8 * kh;
          if (key > qrow) sacc[j][r] = -3.0e38f;
        }
      }
    }

    // online softmax: reduce across 16-lane halves via immediate ds_swizzle
    #pragma unroll
    for (int r = 0; r < 8; ++r) {
      float rmax = -3.0e38f;
      #pragma unroll
      for (int j = 0; j < 4; ++j) rmax = fmaxf(rmax, sacc[j][r] * scale);
      rmax = fmaxf(rmax, SWZF(rmax, 1));
      rmax = fmaxf(rmax, SWZF(rmax, 2));
      rmax = fmaxf(rmax, SWZF(rmax, 4));
      rmax = fmaxf(rmax, SWZF(rmax, 8));
      float mnew = fmaxf(mi[r], rmax);
      float corr = __expf(mi[r] - mnew);
      float rsum = 0.0f;
      #pragma unroll
      for (int j = 0; j < 4; ++j) {
        float p = __expf(sacc[j][r] * scale - mnew);
        sacc[j][r] = p;
        rsum += p;
      }
      rsum += SWZF(rsum, 1);
      rsum += SWZF(rsum, 2);
      rsum += SWZF(rsum, 4);
      rsum += SWZF(rsum, 8);
      li[r] = li[r] * corr + rsum;
      mi[r] = mnew;
      #pragma unroll
      for (int n = 0; n < 4; ++n) oacc[n][r] *= corr;
    }

    // P -> LDS (D-layout -> A-layout; wave-private band, in-order LDS)
    #pragma unroll
    for (int j = 0; j < 4; ++j)
      #pragma unroll
      for (int r = 0; r < 8; ++r)
        Ps[wave * 16 + r + 8 * kh][j * 16 + l15] = (_Float16)sacc[j][r];

    // O += P @ V
    AFrag pa[2];
    {
      int prow = wave * 16 + l15;
      #pragma unroll
      for (int s = 0; s < 2; ++s) {
        pa[s].h[0] = *reinterpret_cast<const v8h*>(&Ps[prow][s * 32 + kh * 8]);
        pa[s].h[1] = *reinterpret_cast<const v8h*>(&Ps[prow][s * 32 + 16 + kh * 8]);
      }
    }
    #pragma unroll
    for (int n = 0; n < 4; ++n) {
      int hs = n * 16 + l15;
      AFrag bv0, bv1;
      bv0.h[0] = *reinterpret_cast<const v8h*>(&Vt[hs][kh * 8]);
      bv0.h[1] = *reinterpret_cast<const v8h*>(&Vt[hs][16 + kh * 8]);
      bv1.h[0] = *reinterpret_cast<const v8h*>(&Vt[hs][32 + kh * 8]);
      bv1.h[1] = *reinterpret_cast<const v8h*>(&Vt[hs][48 + kh * 8]);
      oacc[n] = __builtin_amdgcn_wmma_f32_16x16x32_f16(
          false, pa[0].v, false, bv0.v, (short)0, oacc[n], false, false);
      oacc[n] = __builtin_amdgcn_wmma_f32_16x16x32_f16(
          false, pa[1].v, false, bv1.v, (short)0, oacc[n], false, false);
    }
  }

  #pragma unroll
  for (int r = 0; r < 8; ++r) {
    float inv = 1.0f / li[r];
    int row = q0 + wave * 16 + r + 8 * kh;
    _Float16* yp = Y + bOff + (size_t)row * EMB;
    #pragma unroll
    for (int n = 0; n < 4; ++n)
      yp[n * 16 + l15] = (_Float16)(oacc[n][r] * inv);
  }
}

extern "C" void kernel_launch(void* const* d_in, const int* in_sizes, int n_in,
                              void* d_out, int out_size, void* d_ws, size_t ws_size,
                              hipStream_t stream) {
  const float* x  = (const float*)d_in[0];
  const float* Wq = (const float*)d_in[1];
  const float* bq = (const float*)d_in[2];
  const float* Wk = (const float*)d_in[3];
  const float* bk = (const float*)d_in[4];
  const float* Wv = (const float*)d_in[5];
  const float* bv = (const float*)d_in[6];
  const float* Wo = (const float*)d_in[7];
  const float* bo = (const float*)d_in[8];
  float* out = (float*)d_out;

  const int M = 4 * 2048, N = 1024, Kd = 1024;
  const size_t MK = (size_t)M * Kd, KN = (size_t)Kd * N;

  _Float16* xh  = (_Float16*)d_ws;
  _Float16* Wqt = xh  + MK;
  _Float16* Wkt = Wqt + KN;
  _Float16* Wvt = Wkt + KN;
  _Float16* Wot = Wvt + KN;
  _Float16* q   = Wot + KN;
  _Float16* k   = q + MK;
  _Float16* v   = k + MK;
  _Float16* y   = v + MK;   // total: 16 + 8 + 64 MB

  cvt_f32_f16_kernel<<<(int)(MK / 4 / 256), 256, 0, stream>>>(x, xh, (int)MK);

  dim3 tb(32, 8), tg(N / 32, Kd / 32);
  transpose_cvt_kernel<<<tg, tb, 0, stream>>>(Wq, Wqt, Kd, N);
  transpose_cvt_kernel<<<tg, tb, 0, stream>>>(Wk, Wkt, Kd, N);
  transpose_cvt_kernel<<<tg, tb, 0, stream>>>(Wv, Wvt, Kd, N);
  transpose_cvt_kernel<<<tg, tb, 0, stream>>>(Wo, Wot, Kd, N);

  dim3 gg(N / 256, M / 128);
  gemm_tdm_kernel<_Float16><<<gg, 256, 0, stream>>>(xh, Wqt, bq, q, M, N, Kd);
  gemm_tdm_kernel<_Float16><<<gg, 256, 0, stream>>>(xh, Wkt, bk, k, M, N, Kd);
  gemm_tdm_kernel<_Float16><<<gg, 256, 0, stream>>>(xh, Wvt, bv, v, M, N, Kd);

  flash_attn_kernel<<<dim3(2048 / 64, 16, 4), 128, 0, stream>>>(q, k, v, y);

  gemm_tdm_kernel<float><<<gg, 256, 0, stream>>>(y, Wot, bo, out, M, N, Kd);
}
